// MSCAT_Surv_86792699118199
// MI455X (gfx1250) — matrix operand order repaired and checked by reading, not compiled, not run on hardware
//
#include <hip/hip_runtime.h>

typedef __attribute__((ext_vector_type(16))) __bf16 v16bf;
typedef __attribute__((ext_vector_type(8)))  __bf16 v8bf;
typedef __attribute__((ext_vector_type(8)))  float  v8f;
typedef __attribute__((ext_vector_type(4)))  unsigned int v4u;
typedef __attribute__((ext_vector_type(4)))  int    v4i;
typedef __attribute__((ext_vector_type(8)))  int    v8i;

constexpr int   kS     = 2;
constexpr int   kD     = 1024;
constexpr int   kB     = 8192;
constexpr int   kM     = 4096;
constexpr float kEPS   = 1e-5f;
constexpr float kSCALE = 0.03125f;  // 1/sqrt(1024)
constexpr float kDECAY = 0.1f;

enum { LB_NN = 0, LB_NT = 1 };
enum { EPI_PLAIN = 0, EPI_SCALE_DECAY = 1, EPI_RELU_BF16 = 2, EPI_ADD_DUAL = 3, EPI_ADD_RELU = 4 };

// ---------------------------------------------------------------------------
// TDM: load a (rows x 32) bf16 tile (row stride = stride_elems) into LDS.
// D# per ISA 8.3/8.4: count=1, type=2, data_size=1 (2B), pad_enable with
// interval=16 DWORDs (one 64B row) and amount=4 DWORDs (16B) -> LDS rows of
// 40 halves (80B, 16B-aligned, conflict-free for the fragment loads below).
// ---------------------------------------------------------------------------
__device__ __forceinline__ void tdm_load_tile(const __bf16* gsrc, unsigned lds_off,
                                              unsigned tile_rows, long long stride_elems) {
  unsigned long long ga = (unsigned long long)(size_t)gsrc;
  v4u g0;
  g0[0] = 1u;                                                  // count=1 (valid)
  g0[1] = lds_off;                                             // lds_addr (bytes)
  g0[2] = (unsigned)(ga & 0xFFFFFFFFu);                        // global_addr[31:0]
  g0[3] = (unsigned)((ga >> 32) & 0x01FFFFFFu) | (2u << 30);   // addr[56:32], type=2
  const unsigned flags = (1u << 16)    // data_size = 1 -> 2 bytes
                       | (1u << 20)    // pad_enable
                       | (3u << 22)    // pad_interval code 3 -> 16 DWORDs
                       | (3u << 25);   // pad_amount  code 3 -> 4 DWORDs
  const unsigned td0 = 0x7FFFFFFFu, td1 = 0x7FFFFFFFu;         // no OOB clipping
  const unsigned tile0 = 32u;
  unsigned long long s0 = (unsigned long long)stride_elems;
  v8i g1;
  g1[0] = (int)flags;
  g1[1] = (int)((td0 & 0xFFFFu) << 16);                        // tensor_dim0 lo16
  g1[2] = (int)((td0 >> 16) | ((td1 & 0xFFFFu) << 16));        // dim0 hi | dim1 lo
  g1[3] = (int)((td1 >> 16) | (tile0 << 16));                  // dim1 hi | tile_dim0
  g1[4] = (int)(tile_rows & 0xFFFFu);                          // tile_dim1 (tile_dim2=0)
  g1[5] = (int)(s0 & 0xFFFFFFFFu);                             // dim0_stride lo
  g1[6] = (int)((s0 >> 32) & 0xFFFFu);                         // dim0_stride hi
  g1[7] = 0;
  v4i z4 = {0, 0, 0, 0};
#if __clang_major__ >= 23
  v8i z8 = {0, 0, 0, 0, 0, 0, 0, 0};
  __builtin_amdgcn_tensor_load_to_lds(g0, g1, z4, z4, z8, 0);
#else
  __builtin_amdgcn_tensor_load_to_lds(g0, g1, z4, z4, 0);
#endif
}

// ---------------------------------------------------------------------------
// Stage one K-step (32) of tiles into LDS buffers.
//  A: 128 x 32 bf16, via TDM (wave 0).
//  B (NT, source N x K): 64 x 32 bf16, via TDM (wave 1) -> Bs[n][k].
//  B (NN, source K x N): manual transpose staging -> Bs[n][k].
// ---------------------------------------------------------------------------
template <int BLAY>
__device__ __forceinline__ void stage_tiles(const __bf16* Aop, const __bf16* Bop,
                                            int K, int ldb, int m0, int n0, int k0,
                                            __bf16 (*Asb)[40], __bf16 (*Bsb)[40],
                                            int tid, int wave) {
  if (wave == 0)
    tdm_load_tile(Aop + (size_t)m0 * K + k0, (unsigned)(size_t)&Asb[0][0], 128u, K);
  if constexpr (BLAY == LB_NT) {
    if (wave == 1)
      tdm_load_tile(Bop + (size_t)n0 * ldb + k0, (unsigned)(size_t)&Bsb[0][0], 64u, ldb);
  } else {
    const int k  = tid >> 3;           // 0..31
    const int nc = (tid & 7) * 8;      // 0..56
    const v8bf vv = *(const v8bf*)(Bop + (size_t)(k0 + k) * ldb + n0 + nc);
#pragma unroll
    for (int j = 0; j < 8; ++j) Bsb[nc + j][k] = vv[j];
  }
}

// ---------------------------------------------------------------------------
// WMMA GEMM: out[m,n] = epi( sum_k A[m,k] * B(k,n) + bias[n] )
// Block tile 128x64, 8 waves as 4(M) x 2(N), each wave 32x32 (4 accumulators,
// 4 x v_wmma_f32_16x16x32_bf16 per K-step). Double-buffered LDS, TDM staged.
// ---------------------------------------------------------------------------
template <int BLAY, int EPI>
__global__ __launch_bounds__(256) void gemm_wmma_kernel(
    const __bf16* __restrict__ Aop, const __bf16* __restrict__ Bop,
    const float* __restrict__ bias, const float* __restrict__ addsrc,
    float* __restrict__ outf, __bf16* __restrict__ outb,
    int Mrows, int Ncols, int K)
{
  __shared__ __align__(16) __bf16 As[2][128][40];
  __shared__ __align__(16) __bf16 Bs[2][64][40];

  const int tid  = threadIdx.x;
  const int m0   = blockIdx.y * 128;
  const int n0   = blockIdx.x * 64;
  const int lane = tid & 31;           // wave32
  const int wave = tid >> 5;
  const int wm   = wave >> 1;          // 0..3
  const int wn   = wave & 1;           // 0..1
  const int g    = lane >> 4;          // 0..1
  const int ln   = lane & 15;          // 0..15

  const int ldb = (BLAY == LB_NN) ? Ncols : K;
  const int nk  = K >> 5;

  v8f acc[2][2];
#pragma unroll
  for (int i = 0; i < 2; ++i)
#pragma unroll
    for (int j = 0; j < 2; ++j) acc[i][j] = (v8f){0.f,0.f,0.f,0.f,0.f,0.f,0.f,0.f};

  stage_tiles<BLAY>(Aop, Bop, K, ldb, m0, n0, 0, As[0], Bs[0], tid, wave);

  for (int kt = 0; kt < nk; ++kt) {
    const int cur = kt & 1;
    if (kt + 1 < nk) {
      stage_tiles<BLAY>(Aop, Bop, K, ldb, m0, n0, (kt + 1) << 5,
                        As[cur ^ 1], Bs[cur ^ 1], tid, wave);
      __builtin_amdgcn_s_wait_tensorcnt(1);   // tile kt landed; kt+1 in flight
    } else {
      __builtin_amdgcn_s_wait_tensorcnt(0);
    }
    __syncthreads();

    // Fragment loads: all 16B-aligned contiguous -> ds_load_b128 pairs.
    const __bf16* arow = &As[cur][wm * 32 + ln][0];
    const __bf16* brow = &Bs[cur][wn * 32 + ln][0];
    v8bf a0lo = *(const v8bf*)(arow + g * 8);
    v8bf a0hi = *(const v8bf*)(arow + 16 + g * 8);
    v8bf a1lo = *(const v8bf*)(arow + 16 * 40 + g * 8);
    v8bf a1hi = *(const v8bf*)(arow + 16 * 40 + 16 + g * 8);
    v8bf b0lo = *(const v8bf*)(brow + g * 8);
    v8bf b0hi = *(const v8bf*)(brow + 16 + g * 8);
    v8bf b1lo = *(const v8bf*)(brow + 16 * 40 + g * 8);
    v8bf b1hi = *(const v8bf*)(brow + 16 * 40 + 16 + g * 8);
    v16bf a0 = __builtin_shufflevector(a0lo, a0hi, 0,1,2,3,4,5,6,7,8,9,10,11,12,13,14,15);
    v16bf a1 = __builtin_shufflevector(a1lo, a1hi, 0,1,2,3,4,5,6,7,8,9,10,11,12,13,14,15);
    v16bf b0 = __builtin_shufflevector(b0lo, b0hi, 0,1,2,3,4,5,6,7,8,9,10,11,12,13,14,15);
    v16bf b1 = __builtin_shufflevector(b1lo, b1hi, 0,1,2,3,4,5,6,7,8,9,10,11,12,13,14,15);

    acc[0][0] = __builtin_amdgcn_wmma_f32_16x16x32_bf16(false, a0, false, b0, (short)0, acc[0][0], false, false);
    acc[0][1] = __builtin_amdgcn_wmma_f32_16x16x32_bf16(false, a0, false, b1, (short)0, acc[0][1], false, false);
    acc[1][0] = __builtin_amdgcn_wmma_f32_16x16x32_bf16(false, a1, false, b0, (short)0, acc[1][0], false, false);
    acc[1][1] = __builtin_amdgcn_wmma_f32_16x16x32_bf16(false, a1, false, b1, (short)0, acc[1][1], false, false);
    __syncthreads();
  }

  // Epilogue; C/D layout: VGPR r -> M = g*8 + r (lanes 16-31: +8 handled by g), N = ln.
  const int gnb = n0 + wn * 32 + ln;
  const float bb0 = bias ? bias[gnb] : 0.f;
  const float bb1 = bias ? bias[gnb + 16] : 0.f;
#pragma unroll
  for (int i = 0; i < 2; ++i) {
#pragma unroll
    for (int j = 0; j < 2; ++j) {
      const int gn = gnb + j * 16;
      const float bb = j ? bb1 : bb0;
#pragma unroll
      for (int r = 0; r < 8; ++r) {
        const int gm = m0 + wm * 32 + i * 16 + g * 8 + r;
        const size_t idx = (size_t)gm * Ncols + gn;
        float v = acc[i][j][r] + bb;
        if constexpr (EPI == EPI_PLAIN) {
          outf[idx] = v;
        } else if constexpr (EPI == EPI_SCALE_DECAY) {
          outf[idx] = v * kSCALE - kDECAY * (float)gm;
        } else if constexpr (EPI == EPI_RELU_BF16) {
          outb[idx] = (__bf16)fmaxf(v, 0.f);
        } else if constexpr (EPI == EPI_ADD_DUAL) {
          const float r2 = addsrc[idx] + v;
          outf[idx] = r2;
          outb[idx] = (__bf16)r2;
        } else if constexpr (EPI == EPI_ADD_RELU) {
          outf[idx] = addsrc[idx] + fmaxf(v, 0.f);
        }
      }
    }
  }
}

// ---------------------------------------------------------------------------
// Elementwise / reduction kernels (f32 math, optional bf16 sidecar outputs)
// ---------------------------------------------------------------------------
__global__ void cvt_bf16_kernel(const float* __restrict__ in, __bf16* __restrict__ out, int n) {
  int i = blockIdx.x * blockDim.x + threadIdx.x;
  if (i < n) out[i] = (__bf16)in[i];
}

__global__ void add_pe0_kernel(const float* __restrict__ in, float* __restrict__ outf,
                               __bf16* __restrict__ outb, int n) {
  int i = blockIdx.x * blockDim.x + threadIdx.x;
  if (i >= n) return;
  const float v = in[i] + ((i & 1) ? 1.0f : 0.0f);
  if (outf) outf[i] = v;
  if (outb) outb[i] = (__bf16)v;
}

__global__ __launch_bounds__(256) void layernorm_kernel(
    const float* __restrict__ in, float* __restrict__ outf, __bf16* __restrict__ outb,
    const float* __restrict__ gma, const float* __restrict__ bta)
{
  const int row = blockIdx.x;
  const int t   = threadIdx.x;
  const float* rp = in + (size_t)row * kD;
  __shared__ float red[256];
  float v[4];
  float s = 0.f;
#pragma unroll
  for (int j = 0; j < 4; ++j) { v[j] = rp[t + j * 256]; s += v[j]; }
  red[t] = s; __syncthreads();
  for (int o = 128; o > 0; o >>= 1) { if (t < o) red[t] += red[t + o]; __syncthreads(); }
  const float mu = red[0] * (1.0f / kD);
  __syncthreads();
  float s2 = 0.f;
#pragma unroll
  for (int j = 0; j < 4; ++j) { float d = v[j] - mu; s2 += d * d; }
  red[t] = s2; __syncthreads();
  for (int o = 128; o > 0; o >>= 1) { if (t < o) red[t] += red[t + o]; __syncthreads(); }
  const float inv = rsqrtf(red[0] * (1.0f / kD) + kEPS);
#pragma unroll
  for (int j = 0; j < 4; ++j) {
    const int c = t + j * 256;
    const float o = (v[j] - mu) * inv * gma[c] + bta[c];
    if (outf) outf[(size_t)row * kD + c] = o;
    if (outb) outb[(size_t)row * kD + c] = (__bf16)o;
  }
}

__global__ void rope_kernel(const float* __restrict__ in, __bf16* __restrict__ out, int npairs) {
  int i = blockIdx.x * blockDim.x + threadIdx.x;
  if (i >= npairs) return;
  const int m = i >> 9;          // D/2 = 512 pairs per row
  const int p = i & 511;
  const float ang = (float)m * __powf(10000.0f, -(float)p * (1.0f / 511.0f));
  const float sn = __sinf(ang);
  const float cs = __cosf(ang);
  const float* rp = in + (size_t)m * kD + 2 * p;
  const float a = rp[0], b = rp[1];
  out[(size_t)m * kD + 2 * p]     = (__bf16)(a * cs - b * sn);
  out[(size_t)m * kD + 2 * p + 1] = (__bf16)(b * cs + a * sn);
}

__global__ __launch_bounds__(256) void softmax_kernel(
    const float* __restrict__ in, float* __restrict__ outf, __bf16* __restrict__ outb)
{
  const int row = blockIdx.x;
  const int t   = threadIdx.x;
  const float* rp = in + (size_t)row * kB;
  __shared__ float red[256];
  float v[32];
  float mx = -3.4e38f;
#pragma unroll
  for (int j = 0; j < 32; ++j) { v[j] = rp[t + j * 256]; mx = fmaxf(mx, v[j]); }
  red[t] = mx; __syncthreads();
  for (int o = 128; o > 0; o >>= 1) { if (t < o) red[t] = fmaxf(red[t], red[t + o]); __syncthreads(); }
  const float rmax = red[0];
  __syncthreads();
  float s = 0.f;
#pragma unroll
  for (int j = 0; j < 32; ++j) { v[j] = __expf(v[j] - rmax); s += v[j]; }
  red[t] = s; __syncthreads();
  for (int o = 128; o > 0; o >>= 1) { if (t < o) red[t] += red[t + o]; __syncthreads(); }
  const float inv = 1.0f / red[0];
#pragma unroll
  for (int j = 0; j < 32; ++j) {
    const float p = v[j] * inv;
    outf[(size_t)row * kB + t + j * 256] = p;
    outb[(size_t)row * kB + t + j * 256] = (__bf16)p;
  }
}

__global__ void concat_kernel(const float* __restrict__ hnew, const float* __restrict__ hin,
                              __bf16* __restrict__ cat, int n) {
  int i = blockIdx.x * blockDim.x + threadIdx.x;
  if (i >= n) return;
  const int row = i / (2 * kD);
  const int col = i - row * (2 * kD);
  const float v = (col < kD) ? hnew[(size_t)row * kD + col] : hin[(size_t)row * kD + col - kD];
  cat[i] = (__bf16)v;
}

__global__ void blend_kernel(const float* __restrict__ gate, const float* __restrict__ hnew,
                             const float* __restrict__ hin, float* __restrict__ houtf,
                             __bf16* __restrict__ houtb, int n) {
  int i = blockIdx.x * blockDim.x + threadIdx.x;
  if (i >= n) return;
  const float gg = 1.0f / (1.0f + __expf(-gate[i]));
  const float v = gg * hnew[i] + (1.0f - gg) * hin[i];
  houtf[i] = v;
  houtb[i] = (__bf16)v;
}

// ---------------------------------------------------------------------------
extern "C" void kernel_launch(void* const* d_in, const int* in_sizes, int n_in,
                              void* d_out, int out_size, void* d_ws, size_t ws_size,
                              hipStream_t stream) {
  (void)in_sizes; (void)n_in; (void)out_size; (void)ws_size;

  const float* x      = (const float*)d_in[0];
  const float* y      = (const float*)d_in[1];
  const float* attn_w = (const float*)d_in[2];
  const float* attn_b = (const float*)d_in[3];
  const float* ln_g   = (const float*)d_in[4];
  const float* ln_b   = (const float*)d_in[5];
  const float* gate_w = (const float*)d_in[6];
  const float* gate_b = (const float*)d_in[7];
  const float* res_w  = (const float*)d_in[8];
  const float* res_b  = (const float*)d_in[9];
  const float* ffn_w1 = (const float*)d_in[10];
  const float* ffn_b1 = (const float*)d_in[11];
  const float* ffn_w2 = (const float*)d_in[12];
  const float* ffn_b2 = (const float*)d_in[13];
  const float* fln_g  = (const float*)d_in[14];
  const float* fln_b  = (const float*)d_in[15];

  float* outp = (float*)d_out;                 // [M*D] h, then [M*B] A
  float* outA = outp + (size_t)kM * kD;

  // --- f32 workspace ---
  float* pf = (float*)d_ws;
  float* hf0   = pf; pf += (size_t)kM * kD;
  float* hf1   = pf; pf += (size_t)kM * kD;
  float* xtf   = pf; pf += (size_t)kB * kD;
  float* ytf   = pf; pf += (size_t)kM * kD;
  float* sc    = pf; pf += (size_t)kM * kB;
  float* hnew  = pf; pf += (size_t)kM * kD;
  float* gateb = pf; pf += (size_t)kM * kD;
  // --- bf16 workspace ---
  __bf16* pb = (__bf16*)pf;
  __bf16* x2b  = pb; pb += (size_t)kB * kD;
  __bf16* hb0  = pb; pb += (size_t)kM * kD;
  __bf16* hb1  = pb; pb += (size_t)kM * kD;
  __bf16* xtb  = pb; pb += (size_t)kB * kD;
  __bf16* ytb  = pb; pb += (size_t)kM * kD;
  __bf16* Ab   = pb; pb += (size_t)kM * kB;
  __bf16* catb = pb; pb += (size_t)kM * 2 * kD;
  __bf16* t1b  = pb; pb += (size_t)kM * 4 * kD;
  __bf16* awb  = pb; pb += (size_t)kS * kD * kD;
  __bf16* gwb  = pb; pb += (size_t)kS * kD * 2 * kD;
  __bf16* rwb  = pb; pb += (size_t)kS * kD * kD;
  __bf16* f1b  = pb; pb += (size_t)4 * kD * kD;
  __bf16* f2b  = pb; pb += (size_t)kD * 4 * kD;

  // Weights -> bf16 once per call.
  cvt_bf16_kernel<<<(kS * kD * kD + 255) / 256, 256, 0, stream>>>(attn_w, awb, kS * kD * kD);
  cvt_bf16_kernel<<<(kS * kD * 2 * kD + 255) / 256, 256, 0, stream>>>(gate_w, gwb, kS * kD * 2 * kD);
  cvt_bf16_kernel<<<(kS * kD * kD + 255) / 256, 256, 0, stream>>>(res_w, rwb, kS * kD * kD);
  cvt_bf16_kernel<<<(4 * kD * kD + 255) / 256, 256, 0, stream>>>(ffn_w1, f1b, 4 * kD * kD);
  cvt_bf16_kernel<<<(4 * kD * kD + 255) / 256, 256, 0, stream>>>(ffn_w2, f2b, 4 * kD * kD);

  // x2 = x + pe0 (bf16 only, GEMM A operand); h0 = y + pe0 (f32 + bf16).
  add_pe0_kernel<<<(kB * kD + 255) / 256, 256, 0, stream>>>(x, nullptr, x2b, kB * kD);
  add_pe0_kernel<<<(kM * kD + 255) / 256, 256, 0, stream>>>(y, hf0, hb0, kM * kD);

  float*  hf[2] = {hf0, hf1};
  __bf16* hb[2] = {hb0, hb1};

  for (int i = 0; i < kS; ++i) {
    float*  hin_f  = hf[i & 1];
    __bf16* hin_b  = hb[i & 1];
    float*  hout_f = hf[(i + 1) & 1];
    __bf16* hout_b = hb[(i + 1) & 1];
    const __bf16* aw = awb + (size_t)i * kD * kD;
    const float*  ab = attn_b + (size_t)i * kD;

    // xt = LN(x2 @ aw.T + ab)  (bf16 only; consumed as B operand)
    gemm_wmma_kernel<LB_NT, EPI_PLAIN><<<dim3(kD / 64, kB / 128), 256, 0, stream>>>(
        x2b, aw, ab, nullptr, xtf, nullptr, kB, kD, kD);
    layernorm_kernel<<<kB, 256, 0, stream>>>(xtf, nullptr, xtb,
        ln_g + (size_t)i * kD, ln_b + (size_t)i * kD);

    // yt = rope(LN(hin @ aw.T + ab))  (bf16 only; GEMM A operand)
    gemm_wmma_kernel<LB_NT, EPI_PLAIN><<<dim3(kD / 64, kM / 128), 256, 0, stream>>>(
        hin_b, aw, ab, nullptr, ytf, nullptr, kM, kD, kD);
    layernorm_kernel<<<kM, 256, 0, stream>>>(ytf, ytf, nullptr,
        ln_g + (size_t)i * kD, ln_b + (size_t)i * kD);
    rope_kernel<<<(kM * kD / 2 + 255) / 256, 256, 0, stream>>>(ytf, ytb, kM * kD / 2);

    // scores = yt @ xt.T * SCALE - m*DECAY ; A = softmax(scores)
    gemm_wmma_kernel<LB_NT, EPI_SCALE_DECAY><<<dim3(kB / 64, kM / 128), 256, 0, stream>>>(
        ytb, xtb, nullptr, nullptr, sc, nullptr, kM, kB, kD);
    float* Abuf = (i == kS - 1) ? outA : sc;
    softmax_kernel<<<kM, 256, 0, stream>>>(sc, Abuf, Ab);

    // h_new = A @ xt   (NN: B operand is K x N row-major)
    gemm_wmma_kernel<LB_NN, EPI_PLAIN><<<dim3(kD / 64, kM / 128), 256, 0, stream>>>(
        Ab, xtb, nullptr, nullptr, hnew, nullptr, kM, kD, kB);

    // gate = sigmoid([h_new, hin] @ gate_w.T + gate_b); hout = blend
    concat_kernel<<<(kM * 2 * kD + 255) / 256, 256, 0, stream>>>(hnew, hin_f, catb, kM * 2 * kD);
    gemm_wmma_kernel<LB_NT, EPI_PLAIN><<<dim3(kD / 64, kM / 128), 256, 0, stream>>>(
        catb, gwb + (size_t)i * kD * 2 * kD, gate_b + (size_t)i * kD, nullptr,
        gateb, nullptr, kM, kD, 2 * kD);
    blend_kernel<<<(kM * kD + 255) / 256, 256, 0, stream>>>(gateb, hnew, hin_f,
        hout_f, hout_b, kM * kD);

    // h += relu(h @ ffn_w1.T + b1) @ ffn_w2.T + b2
    gemm_wmma_kernel<LB_NT, EPI_RELU_BF16><<<dim3(4 * kD / 64, kM / 128), 256, 0, stream>>>(
        hout_b, f1b, ffn_b1, nullptr, nullptr, t1b, kM, 4 * kD, kD);
    gemm_wmma_kernel<LB_NT, EPI_ADD_DUAL><<<dim3(kD / 64, kM / 128), 256, 0, stream>>>(
        t1b, f2b, ffn_b2, hout_f, hout_f, hout_b, kM, kD, 4 * kD);

    // tmp = h + relu(h @ res_w.T + res_b)   (into gateb; avoids A/out alias)
    gemm_wmma_kernel<LB_NT, EPI_ADD_RELU><<<dim3(kD / 64, kM / 128), 256, 0, stream>>>(
        hout_b, rwb + (size_t)i * kD * kD, res_b + (size_t)i * kD, hout_f,
        gateb, nullptr, kM, kD, kD);

    // h = LN(tmp)
    float*  lnf = (i == kS - 1) ? outp : hout_f;
    __bf16* lnb = (i == kS - 1) ? nullptr : hout_b;
    layernorm_kernel<<<kM, 256, 0, stream>>>(gateb, lnf, lnb, fln_g, fln_b);
  }
}
